// MultiHeadAttention_60851096650126
// MI455X (gfx1250) — matrix operand order repaired and checked by reading, not compiled, not run on hardware
//
#include <hip/hip_runtime.h>

// ---------- types ----------
typedef __attribute__((ext_vector_type(16))) __bf16 v16bf;
typedef __attribute__((ext_vector_type(8)))  __bf16 v8bf;
typedef __attribute__((ext_vector_type(4)))  __bf16 v4bf;
typedef __attribute__((ext_vector_type(8)))  float  v8f;
typedef __attribute__((ext_vector_type(4)))  float  v4f;

union FragBF {
    v16bf v;
    struct { v8bf lo, hi; } h;
};

__device__ __forceinline__ v8f wmma_bf16(const FragBF& a, const FragBF& b, v8f c) {
    return __builtin_amdgcn_wmma_f32_16x16x32_bf16(
        /*neg_a=*/false, a.v, /*neg_b=*/false, b.v,
        /*c_mod=*/(short)0, c, /*reuse_a=*/false, /*reuse_b=*/false);
}

// ---------- weight transpose fp32 -> bf16, dst[n][k] = src[k][n], dim=1024 ----------
__global__ __launch_bounds__(256) void transpose_w_bf16(const float* __restrict__ src,
                                                        void* __restrict__ dstv) {
    __bf16* dst = (__bf16*)dstv;
    int idx = blockIdx.x * 256 + threadIdx.x;      // 0 .. 1048575
    int n = idx >> 10, k = idx & 1023;
    dst[idx] = (__bf16)src[k * 1024 + n];
}

// ---------- GEMM: C[8192 x 1024] = A[8192 x 1024] * W^T (W stored [n][k] bf16) + bias ----------
// MODE 0: A fp32 -> q bf16  [(b*16+h)*2048+s]*64+d
// MODE 1: A fp32 -> k bf16  same layout
// MODE 2: A fp32 -> vT bf16 [(b*16+h)*64+d]*2048+s   (pre-transposed for flash P@V)
// MODE 3: A bf16 -> out fp32 row-major [row*1024+col]
template<int MODE>
__global__ __launch_bounds__(256) void gemm_bf16(const void* __restrict__ Av,
                                                 const void* __restrict__ Wv,
                                                 const float* __restrict__ bias,
                                                 void* __restrict__ Ov) {
    constexpr int K = 1024;
    __shared__ __align__(16) char sA[128 * 80];   // 128 rows x 32 bf16, 80B padded stride
    __shared__ __align__(16) char sB[256 * 80];   // 256 n-rows x 32 bf16

    const int tid  = threadIdx.x;
    const int lane = tid & 31, wid = tid >> 5;
    const int ln   = lane & 15, hl  = lane >> 4;
    const int wM   = wid >> 2,  wN  = wid & 3;     // 2 x 4 wave grid, 64x64 per wave
    const int m0   = blockIdx.x * 128;
    const int n0   = blockIdx.y * 256;
    const __bf16* W = (const __bf16*)Wv;

    v8f acc[4][4];
    #pragma unroll
    for (int i = 0; i < 4; ++i)
        #pragma unroll
        for (int j = 0; j < 4; ++j) acc[i][j] = (v8f)0.0f;

    for (int kt = 0; kt < K / 32; ++kt) {
        const int k0 = kt * 32;
        __syncthreads();
        // ---- stage A tile (128 x 32) ----
        if constexpr (MODE < 3) {
            const float* A = (const float*)Av;
            #pragma unroll
            for (int i = 0; i < 4; ++i) {
                int idx = tid + i * 256;          // 0..1023, chunks of 4 fp32
                int r = idx >> 3, c4 = idx & 7;
                v4f d = *(const v4f*)(A + (size_t)(m0 + r) * K + k0 + c4 * 4);
                v4bf bd;
                bd[0] = (__bf16)d[0]; bd[1] = (__bf16)d[1];
                bd[2] = (__bf16)d[2]; bd[3] = (__bf16)d[3];
                *(v4bf*)(sA + r * 80 + c4 * 8) = bd;
            }
        } else {
            const __bf16* A = (const __bf16*)Av;
            #pragma unroll
            for (int i = 0; i < 2; ++i) {
                int idx = tid + i * 256;          // 0..511, chunks of 8 bf16
                int r = idx >> 2, c8 = idx & 3;
                v8bf d = *(const v8bf*)(A + (size_t)(m0 + r) * K + k0 + c8 * 8);
                *(v8bf*)(sA + r * 80 + c8 * 16) = d;
            }
        }
        // ---- stage B tile (256 n x 32 k), W already [n][k] bf16 ----
        #pragma unroll
        for (int i = 0; i < 4; ++i) {
            int idx = tid + i * 256;              // 0..1023
            int n = idx >> 2, c8 = idx & 3;
            v8bf d = *(const v8bf*)(W + (size_t)(n0 + n) * K + k0 + c8 * 8);
            *(v8bf*)(sB + n * 80 + c8 * 16) = d;
        }
        __syncthreads();

        // ---- fragments ----
        FragBF fa[4], fb[4];
        #pragma unroll
        for (int mt = 0; mt < 4; ++mt) {
            const char* ar = sA + (wM * 64 + mt * 16 + ln) * 80;
            fa[mt].h.lo = *(const v8bf*)(ar + hl * 16);        // K = hl*8 + j
            fa[mt].h.hi = *(const v8bf*)(ar + 32 + hl * 16);   // K = 16 + hl*8 + j
        }
        #pragma unroll
        for (int nt = 0; nt < 4; ++nt) {
            const char* br = sB + (wN * 64 + nt * 16 + ln) * 80;
            fb[nt].h.lo = *(const v8bf*)(br + hl * 16);        // K = hl*8 + j
            fb[nt].h.hi = *(const v8bf*)(br + 32 + hl * 16);
        }
        #pragma unroll
        for (int mt = 0; mt < 4; ++mt)
            #pragma unroll
            for (int nt = 0; nt < 4; ++nt)
                acc[mt][nt] = wmma_bf16(fa[mt], fb[nt], acc[mt][nt]);
    }

    // ---- epilogue ----
    #pragma unroll
    for (int nt = 0; nt < 4; ++nt) {
        const int col = n0 + wN * 64 + nt * 16 + ln;
        const float bv = bias[col];
        #pragma unroll
        for (int mt = 0; mt < 4; ++mt) {
            #pragma unroll
            for (int v = 0; v < 8; ++v) {
                const int row = m0 + wM * 64 + mt * 16 + hl * 8 + v;
                const float val = acc[mt][nt][v] + bv;
                if constexpr (MODE == 3) {
                    ((float*)Ov)[(size_t)row * 1024 + col] = val;
                } else {
                    const int b = row >> 11, s = row & 2047;
                    const int h = col >> 6,  d = col & 63;
                    __bf16* O = (__bf16*)Ov;
                    if constexpr (MODE == 2)
                        O[((((size_t)b * 16 + h) * 64) + d) * 2048 + s] = (__bf16)val;
                    else
                        O[((((size_t)b * 16 + h) * 2048) + s) * 64 + d] = (__bf16)val;
                }
            }
        }
    }
}

// ---------- flash attention: per (b,h), 128 q rows / block, 4 waves x 32 rows ----------
__global__ __launch_bounds__(128) void attn_flash(const void* __restrict__ qv,
                                                  const void* __restrict__ kv,
                                                  const void* __restrict__ vv,
                                                  void* __restrict__ ov) {
    __shared__ __align__(16) char smem[36864];
    char* kbuf = smem;                 // 64 keys x 64 bf16, 144B stride  (9216B)
    char* vbuf = smem + 9216;          // 64 dk   x 64 bf16, 144B stride  (9216B)
    const int tid  = threadIdx.x;
    const int lane = tid & 31, wid = tid >> 5;
    const int ln   = lane & 15, hl  = lane >> 4;
    char* pbuf = smem + 18432 + wid * 4608;   // wave-private 32 x 64 bf16, 144B stride

    const int bh = blockIdx.y;                // b*16 + h
    const int q0 = blockIdx.x * 128 + wid * 32;
    const __bf16* qm = (const __bf16*)qv + ((size_t)bh * 2048 + q0) * 64;
    const __bf16* km = (const __bf16*)kv + (size_t)bh * 2048 * 64;
    const __bf16* vT = (const __bf16*)vv + (size_t)bh * 64 * 2048;

    // Q fragments resident in registers across whole key loop (32 rows x 64 dk bf16)
    FragBF fq[2][2];
    #pragma unroll
    for (int mt = 0; mt < 2; ++mt)
        #pragma unroll
        for (int kc = 0; kc < 2; ++kc) {
            const __bf16* p = qm + (mt * 16 + ln) * 64 + kc * 32;
            fq[mt][kc].h.lo = *(const v8bf*)(p + hl * 8);
            fq[mt][kc].h.hi = *(const v8bf*)(p + 16 + hl * 8);
        }

    v8f acco[2][4];
    float mst[2][8], lst[2][8];
    #pragma unroll
    for (int mt = 0; mt < 2; ++mt) {
        #pragma unroll
        for (int nt = 0; nt < 4; ++nt) acco[mt][nt] = (v8f)0.0f;
        #pragma unroll
        for (int v = 0; v < 8; ++v) { mst[mt][v] = -3.0e38f; lst[mt][v] = 0.0f; }
    }

    for (int kt = 0; kt < 32; ++kt) {         // 32 key tiles of 64
        const int key0 = kt * 64;
        __syncthreads();
        #pragma unroll
        for (int i = 0; i < 4; ++i) {         // cooperative K / V^T tile staging
            int idx = tid + i * 128;          // 0..511
            int r = idx >> 3, c = idx & 7;
            *(v8bf*)(kbuf + r * 144 + c * 16) =
                *(const v8bf*)(km + (size_t)(key0 + r) * 64 + c * 8);
            *(v8bf*)(vbuf + r * 144 + c * 16) =
                *(const v8bf*)(vT + (size_t)r * 2048 + key0 + c * 8);
        }
        __syncthreads();

        // ---- scores = Q . K^T ----
        v8f sacc[2][4];
        #pragma unroll
        for (int mt = 0; mt < 2; ++mt)
            #pragma unroll
            for (int nt = 0; nt < 4; ++nt) sacc[mt][nt] = (v8f)0.0f;

        #pragma unroll
        for (int nt = 0; nt < 4; ++nt) {      // key n-tiles
            FragBF fk0, fk1;
            const char* kr = kbuf + (nt * 16 + ln) * 144;
            fk0.h.lo = *(const v8bf*)(kr + hl * 16);
            fk0.h.hi = *(const v8bf*)(kr + 32 + hl * 16);
            fk1.h.lo = *(const v8bf*)(kr + 64 + hl * 16);
            fk1.h.hi = *(const v8bf*)(kr + 96 + hl * 16);
            #pragma unroll
            for (int mt = 0; mt < 2; ++mt) {
                sacc[mt][nt] = wmma_bf16(fq[mt][0], fk0, sacc[mt][nt]);
                sacc[mt][nt] = wmma_bf16(fq[mt][1], fk1, sacc[mt][nt]);
            }
        }

        // ---- online softmax (lane-local after half-wave shuffles) ----
        #pragma unroll
        for (int mt = 0; mt < 2; ++mt) {
            #pragma unroll
            for (int v = 0; v < 8; ++v) {
                float t = fmaxf(fmaxf(sacc[mt][0][v], sacc[mt][1][v]),
                                fmaxf(sacc[mt][2][v], sacc[mt][3][v])) * 0.125f;
                #pragma unroll
                for (int off = 8; off; off >>= 1) t = fmaxf(t, __shfl_xor(t, off, 32));
                const float mn = fmaxf(mst[mt][v], t);
                const float al = __expf(mst[mt][v] - mn);
                mst[mt][v] = mn;
                float rs = 0.0f;
                #pragma unroll
                for (int nt = 0; nt < 4; ++nt) {
                    float p = __expf(sacc[mt][nt][v] * 0.125f - mn);
                    sacc[mt][nt][v] = p;
                    rs += p;
                }
                #pragma unroll
                for (int off = 8; off; off >>= 1) rs += __shfl_xor(rs, off, 32);
                lst[mt][v] = lst[mt][v] * al + rs;
                #pragma unroll
                for (int nt = 0; nt < 4; ++nt) acco[mt][nt][v] *= al;
                // P -> wave-private LDS (in-order DS, no barrier needed)
                const int row = mt * 16 + hl * 8 + v;
                #pragma unroll
                for (int nt = 0; nt < 4; ++nt)
                    *(__bf16*)(pbuf + row * 144 + (nt * 16 + ln) * 2) =
                        (__bf16)sacc[mt][nt][v];
            }
        }

        // ---- O += P . V ----
        FragBF fp[2][2];
        #pragma unroll
        for (int mt = 0; mt < 2; ++mt) {
            const char* pr = pbuf + (mt * 16 + ln) * 144;
            fp[mt][0].h.lo = *(const v8bf*)(pr + hl * 16);
            fp[mt][0].h.hi = *(const v8bf*)(pr + 32 + hl * 16);
            fp[mt][1].h.lo = *(const v8bf*)(pr + 64 + hl * 16);
            fp[mt][1].h.hi = *(const v8bf*)(pr + 96 + hl * 16);
        }
        #pragma unroll
        for (int nt = 0; nt < 4; ++nt) {      // dk n-tiles
            FragBF fv0, fv1;
            const char* vr = vbuf + (nt * 16 + ln) * 144;
            fv0.h.lo = *(const v8bf*)(vr + hl * 16);
            fv0.h.hi = *(const v8bf*)(vr + 32 + hl * 16);
            fv1.h.lo = *(const v8bf*)(vr + 64 + hl * 16);
            fv1.h.hi = *(const v8bf*)(vr + 96 + hl * 16);
            #pragma unroll
            for (int mt = 0; mt < 2; ++mt) {
                acco[mt][nt] = wmma_bf16(fp[mt][0], fv0, acco[mt][nt]);
                acco[mt][nt] = wmma_bf16(fp[mt][1], fv1, acco[mt][nt]);
            }
        }
    }

    // ---- finalize: O/l, merged-heads layout (b, s, h*64+d) bf16 ----
    const int b = bh >> 4, h = bh & 15;
    __bf16* O = (__bf16*)ov;
    #pragma unroll
    for (int mt = 0; mt < 2; ++mt)
        #pragma unroll
        for (int v = 0; v < 8; ++v) {
            const float il = 1.0f / lst[mt][v];
            const int s = q0 + mt * 16 + hl * 8 + v;
            #pragma unroll
            for (int nt = 0; nt < 4; ++nt) {
                const int d = nt * 16 + ln;
                O[(((size_t)b * 2048 + s) * 16 + h) * 64 + d] =
                    (__bf16)(acco[mt][nt][v] * il);
            }
        }
}

// ---------- launch ----------
extern "C" void kernel_launch(void* const* d_in, const int* in_sizes, int n_in,
                              void* d_out, int out_size, void* d_ws, size_t ws_size,
                              hipStream_t stream) {
    (void)in_sizes; (void)n_in; (void)out_size; (void)ws_size;
    const float* Q  = (const float*)d_in[0];
    const float* Kk = (const float*)d_in[1];
    const float* V  = (const float*)d_in[2];
    const float* Wq = (const float*)d_in[3];
    const float* bq = (const float*)d_in[4];
    const float* Wo = (const float*)d_in[5];
    const float* bo = (const float*)d_in[6];

    char* ws = (char*)d_ws;
    void* WqT = ws;                          //  2 MB  bf16 Wq^T [n][k]
    void* WoT = ws + (size_t)( 2u << 20);    //  2 MB  bf16 Wo^T [n][k]
    void* qb  = ws + (size_t)( 4u << 20);    // 16 MB  q  bf16 (B,H,S,dk)
    void* kb  = ws + (size_t)(20u << 20);    // 16 MB  k  bf16 (B,H,S,dk)
    void* vb  = ws + (size_t)(36u << 20);    // 16 MB  vT bf16 (B,H,dk,S)
    void* ab  = ws + (size_t)(52u << 20);    // 16 MB  attn out bf16 (B,S,D)

    transpose_w_bf16<<<4096, 256, 0, stream>>>(Wq, WqT);
    transpose_w_bf16<<<4096, 256, 0, stream>>>(Wo, WoT);

    dim3 gg(64, 4);                           // 8192/128 x 1024/256
    gemm_bf16<0><<<gg, 256, 0, stream>>>(Q,  WqT, bq, qb);
    gemm_bf16<1><<<gg, 256, 0, stream>>>(Kk, WqT, bq, kb);
    gemm_bf16<2><<<gg, 256, 0, stream>>>(V,  WqT, bq, vb);

    attn_flash<<<dim3(16, 64), 128, 0, stream>>>(qb, kb, vb, ab);

    gemm_bf16<3><<<gg, 256, 0, stream>>>(ab, WoT, bo, d_out);
}
